// StructureEncoder_36163624632828
// MI455X (gfx1250) — compile-verified, matrix operand order checked by reading, not run on hardware
//
#include <hip/hip_runtime.h>
#include <hip/hip_bf16.h>
#include <stdint.h>

// ---------------------------------------------------------------------------
// StructureEncoder for MI455X (gfx1250, wave32, WMMA).
// Edge/node MLP GEMMs on v_wmma_f32_16x16x32_f16 (f16 in, f32 accum).
// 64-edge M-tiles amortize W1/W2 L2 traffic; x-tile staged with
// GLOBAL_LOAD_ASYNC_TO_LDS_B128 (ASYNCcnt) on gfx1250.
// ---------------------------------------------------------------------------

typedef __attribute__((ext_vector_type(16))) _Float16 v16h;
typedef __attribute__((ext_vector_type(8)))  _Float16 v8h;
typedef __attribute__((ext_vector_type(8)))  float    v8f;
typedef __attribute__((ext_vector_type(4)))  int      v4i;

#define B_      4
#define N_      2048
#define H_      256
#define K_      30
#define NL_     3
#define NRBF    16
#define NNODES  (B_ * N_)          // 8192
#define NEDGES  (NNODES * K_)      // 245760
#define KP1     544                // 2H+16 padded to 32
#define KPU     512                // 2H
#define KP2     256                // H
#define MT      64                 // edge M-tile
#define LN_EPS  1e-5f

// RBF constants: centers linspace(0,20,16) -> spacing 20/15; width 20/16=1.25
#define RBF_CSP   (20.0f / 15.0f)
#define RBF_COEF  (1.0f / (2.0f * 1.25f * 1.25f))

#ifndef __has_builtin
#define __has_builtin(x) 0
#endif

// 16B global -> LDS copy; async (ASYNCcnt) path on gfx1250 toolchains.
__device__ __forceinline__ void cp16_async(void* lds, const void* g) {
#if __has_builtin(__builtin_amdgcn_global_load_async_to_lds_b128)
  typedef __attribute__((address_space(1))) v4i gv4i;   // global int4
  typedef __attribute__((address_space(3))) v4i lv4i;   // LDS int4
  __builtin_amdgcn_global_load_async_to_lds_b128((gv4i*)g, (lv4i*)lds, 0, 0);
#else
  *(uint4*)lds = *(const uint4*)g;
#endif
}

__device__ __forceinline__ void async_join() {
#if __has_builtin(__builtin_amdgcn_global_load_async_to_lds_b128)
#if __has_builtin(__builtin_amdgcn_s_wait_asynccnt)
  __builtin_amdgcn_s_wait_asynccnt(0);
#else
  asm volatile("s_wait_asynccnt 0" ::: "memory");
#endif
#endif
}

__device__ __forceinline__ v16h frag_ld(const _Float16* p) {
  // ISA 16-bit A/B 16x32 layout: per lane two runs of 8 consecutive K halves.
  v8h lo = *(const v8h*)(p);
  v8h hi = *(const v8h*)(p + 16);
  return __builtin_shufflevector(lo, hi, 0,1,2,3,4,5,6,7,8,9,10,11,12,13,14,15);
}

__device__ __forceinline__ float siluf(float x) { return x / (1.0f + __expf(-x)); }

__device__ __forceinline__ v8f wmma16(v16h a, v16h b, v8f c) {
  return __builtin_amdgcn_wmma_f32_16x16x32_f16(false, a, false, b, (short)0, c,
                                                false, false);
}

// --------------------------- weight transpose ------------------------------
// dst[n*Kpad + k] = (f16) src[k*256 + n]; pad k>=K with 0. grid = Kpad x 256thr
__global__ void wtrans(const float* __restrict__ src, _Float16* __restrict__ dst,
                       int K, int Kpad) {
  int idx = blockIdx.x * 256 + threadIdx.x;
  int n = idx / Kpad, k = idx - n * Kpad;
  if (n < 256)
    dst[(size_t)n * Kpad + k] = (_Float16)((k < K) ? src[(size_t)k * 256 + n] : 0.0f);
}

__global__ void zerof(float* __restrict__ p) {
  p[(size_t)blockIdx.x * 256 + threadIdx.x] = 0.0f;
}

// ------------------------------- kNN ---------------------------------------
// one wave per node: dist row in LDS, mean-RBF, 30 selection passes.
__global__ void __launch_bounds__(32) knn(const float* __restrict__ coords,
                                          const unsigned char* __restrict__ mask,
                                          int* __restrict__ nbrflat,
                                          float* __restrict__ rbfmean,
                                          _Float16* __restrict__ ea16,
                                          int* __restrict__ esrc,
                                          int* __restrict__ edst) {
  __shared__ float drow[N_];
  const int flat = blockIdx.x, b = flat >> 11, i = flat & (N_ - 1);
  const int lane = threadIdx.x;
  const float* cb = coords + (size_t)b * N_ * 3;
  const float xi = cb[i * 3 + 0], yi = cb[i * 3 + 1], zi = cb[i * 3 + 2];
  float sum = 0.0f;
  for (int j = lane; j < N_; j += 32) {
    float dx = xi - cb[j * 3 + 0], dy = yi - cb[j * 3 + 1], dz = zi - cb[j * 3 + 2];
    float d = sqrtf(fmaxf(dx * dx + dy * dy + dz * dz, 1e-12f));
    drow[j] = d;
    sum += d;
  }
  __syncthreads();
  #pragma unroll
  for (int m = 16; m >= 1; m >>= 1) sum += __shfl_xor(sum, m, 32);
  const float mean = sum * (1.0f / N_);
  if (lane < NRBF) {
    float df = mean - (float)lane * RBF_CSP;
    rbfmean[(size_t)flat * NRBF + lane] = __expf(-df * df * RBF_COEF);
  }
  const unsigned char mi = mask[flat];
  unsigned long long prev = 0ull;
  for (int s = 0; s < K_; ++s) {
    unsigned long long best = ~0ull;
    for (int j = lane; j < N_; j += 32) {
      float dm = (j == i || !mi || !mask[(size_t)b * N_ + j]) ? 1e10f : drow[j];
      unsigned long long key =
          ((unsigned long long)__float_as_uint(dm) << 32) | (unsigned)j;
      if (key > prev && key < best) best = key;
    }
    #pragma unroll
    for (int m = 16; m >= 1; m >>= 1) {
      unsigned long long o = __shfl_xor(best, m, 32);
      best = (o < best) ? o : best;
    }
    prev = best;
    const int jsel = (int)(best & 0xFFFFFFFFull);
    const float dsel = __uint_as_float((unsigned)(best >> 32));
    const int e = flat * K_ + s;
    if (lane == 0) {
      nbrflat[e] = b * N_ + jsel;
      esrc[e] = flat;
      edst[e] = b * N_ + jsel;
    }
    if (lane < NRBF) {
      float df = dsel - (float)lane * RBF_CSP;
      ea16[(size_t)e * NRBF + lane] = (_Float16)__expf(-df * df * RBF_COEF);
    }
  }
}

// ---------------------------- initial embedding ----------------------------
__global__ void __launch_bounds__(256) init_h(const float* __restrict__ coords,
                                              const unsigned char* __restrict__ mask,
                                              const float* __restrict__ Wa,
                                              const float* __restrict__ ba,
                                              const float* __restrict__ Wp,
                                              const float* __restrict__ bp,
                                              const float* __restrict__ rbfmean,
                                              float* __restrict__ hbuf,
                                              _Float16* __restrict__ hf16) {
  __shared__ float ao[H_];
  const int flat = blockIdx.x, b = flat >> 11, i = flat & (N_ - 1);
  const int t = threadIdx.x;
  const float* cb = coords + (size_t)b * N_ * 3;
  float vpx = 0.f, vpy = 0.f, vpz = 0.f, vsx = 0.f, vsy = 0.f, vsz = 0.f;
  if (i < N_ - 1) {
    float dx = cb[(i + 1) * 3 + 0] - cb[i * 3 + 0];
    float dy = cb[(i + 1) * 3 + 1] - cb[i * 3 + 1];
    float dz = cb[(i + 1) * 3 + 2] - cb[i * 3 + 2];
    float nrm = fmaxf(sqrtf(dx * dx + dy * dy + dz * dz), 1e-12f);
    vpx = dx / nrm; vpy = dy / nrm; vpz = dz / nrm;
  }
  if (i > 0) {
    float dx = cb[i * 3 + 0] - cb[(i - 1) * 3 + 0];
    float dy = cb[i * 3 + 1] - cb[(i - 1) * 3 + 1];
    float dz = cb[i * 3 + 2] - cb[(i - 1) * 3 + 2];
    float nrm = fmaxf(sqrtf(dx * dx + dy * dy + dz * dz), 1e-12f);
    vsx = dx / nrm; vsy = dy / nrm; vsz = dz / nrm;
  }
  float cosv = fminf(fmaxf(vpx * vsx + vpy * vsy + vpz * vsz, -1.0f), 1.0f);
  float sinv = sqrtf(fmaxf(1.0f - cosv * cosv, 0.0f));
  float ac = acosf(cosv);
  float ang[8] = {sinf(ac), cosf(ac), cosv, sinv, 0.f, 0.f, 0.f, 0.f};
  float a = ba[t];
  #pragma unroll
  for (int k = 0; k < 8; ++k) a += ang[k] * Wa[k * H_ + t];
  ao[t] = a;
  __syncthreads();
  float o = bp[t];
  for (int c = 0; c < H_; ++c) o += ao[c] * Wp[(size_t)c * H_ + t];
  #pragma unroll
  for (int r = 0; r < NRBF; ++r)
    o += rbfmean[(size_t)flat * NRBF + r] * Wp[(size_t)(H_ + r) * H_ + t];
  o *= mask[flat] ? 1.0f : 0.0f;
  hbuf[(size_t)flat * H_ + t] = o;
  hf16[(size_t)flat * H_ + t] = (_Float16)o;
}

// ----------------------- edge message + scatter (WMMA) ---------------------
// 64-edge tile per block (dynamic LDS ~100KB), 8 waves x 32 output columns.
__global__ void __launch_bounds__(256) edge_msg(const _Float16* __restrict__ hf16,
                                                const _Float16* __restrict__ ea16,
                                                const int* __restrict__ nbrflat,
                                                const _Float16* __restrict__ Wt1,
                                                const float* __restrict__ b1,
                                                const _Float16* __restrict__ Wt2,
                                                const float* __restrict__ b2,
                                                float* __restrict__ aggr) {
  extern __shared__ __align__(16) char smem_dyn[];
  _Float16* xt = (_Float16*)smem_dyn;                            // [64][544]
  _Float16* y1 = (_Float16*)(smem_dyn + MT * KP1 * 2);           // [64][256]
  int* dsts = (int*)(smem_dyn + MT * KP1 * 2 + MT * KP2 * 2);    // [64]
  const int t = threadIdx.x;
  const int ebase = blockIdx.x * MT;
  {  // stage x = [h[src] | h[dst] | rbf | pad] as f16 (async to LDS)
    const int el = t >> 2, sub = t & 3;  // 4 threads per edge
    const int e = ebase + el;
    const int srcn = e / K_;
    const int dstn = nbrflat[e];
    if (sub == 0) dsts[el] = dstn;
    const uint4* sp = (const uint4*)(hf16 + (size_t)srcn * H_);
    const uint4* dp = (const uint4*)(hf16 + (size_t)dstn * H_);
    uint4* xr = (uint4*)(xt + el * KP1);
    #pragma unroll
    for (int c = sub; c < 32; c += 4) {
      cp16_async(&xr[c], &sp[c]);
      cp16_async(&xr[32 + c], &dp[c]);
    }
    const uint4* ep = (const uint4*)(ea16 + (size_t)e * NRBF);
    if (sub == 0) { cp16_async(&xr[64], &ep[0]); cp16_async(&xr[65], &ep[1]); }
    if (sub == 1) { uint4 z = {0u, 0u, 0u, 0u}; xr[66] = z; xr[67] = z; }
  }
  async_join();
  __syncthreads();

  const int lane = t & 31, n0 = (t >> 5) * 32;
  const int lr = lane & 15, ksel = lane >> 4;
  __builtin_prefetch(Wt2 + (size_t)(n0 + lr) * KP2, 0, 1);  // global_prefetch_b8

  v8f acc[4][2] = {};
  for (int kc = 0; kc < KP1 / 32; ++kc) {
    const int kb = kc * 32 + ksel * 8;
    v16h bf0 = frag_ld(Wt1 + (size_t)(n0 + lr) * KP1 + kb);
    v16h bf1 = frag_ld(Wt1 + (size_t)(n0 + 16 + lr) * KP1 + kb);
    #pragma unroll
    for (int rt = 0; rt < 4; ++rt) {
      v16h a = frag_ld(xt + (size_t)(rt * 16 + lr) * KP1 + kb);
      acc[rt][0] = wmma16(a, bf0, acc[rt][0]);
      acc[rt][1] = wmma16(a, bf1, acc[rt][1]);
    }
  }
  #pragma unroll
  for (int rt = 0; rt < 4; ++rt)
    #pragma unroll
    for (int f = 0; f < 2; ++f) {
      const int col = n0 + f * 16 + lr;
      const float bb = b1[col];
      #pragma unroll
      for (int j = 0; j < 8; ++j) {
        const int row = rt * 16 + j + (ksel ? 8 : 0);
        y1[row * KP2 + col] = (_Float16)siluf(acc[rt][f][j] + bb);
      }
    }
  __syncthreads();

  v8f m[4][2] = {};
  for (int kc = 0; kc < KP2 / 32; ++kc) {
    const int kb = kc * 32 + ksel * 8;
    v16h bf0 = frag_ld(Wt2 + (size_t)(n0 + lr) * KP2 + kb);
    v16h bf1 = frag_ld(Wt2 + (size_t)(n0 + 16 + lr) * KP2 + kb);
    #pragma unroll
    for (int rt = 0; rt < 4; ++rt) {
      v16h a = frag_ld(y1 + (size_t)(rt * 16 + lr) * KP2 + kb);
      m[rt][0] = wmma16(a, bf0, m[rt][0]);
      m[rt][1] = wmma16(a, bf1, m[rt][1]);
    }
  }
  #pragma unroll
  for (int rt = 0; rt < 4; ++rt)
    #pragma unroll
    for (int f = 0; f < 2; ++f) {
      const int col = n0 + f * 16 + lr;
      const float bb = b2[col];
      #pragma unroll
      for (int j = 0; j < 8; ++j) {
        const int row = rt * 16 + j + (ksel ? 8 : 0);
        atomicAdd(&aggr[(size_t)dsts[row] * H_ + col], m[rt][f][j] + bb);
      }
    }
}

// -------------------- node update MLP + residual + LN (WMMA) ---------------
__global__ void __launch_bounds__(256) node_update(float* __restrict__ hbuf,
                                                   _Float16* __restrict__ hf16,
                                                   float* __restrict__ aggr,
                                                   const _Float16* __restrict__ Wu1,
                                                   const float* __restrict__ bu1,
                                                   const _Float16* __restrict__ Wu2,
                                                   const float* __restrict__ bu2,
                                                   const float* __restrict__ lnS,
                                                   const float* __restrict__ lnB,
                                                   const unsigned char* __restrict__ mask) {
  __shared__ __align__(16) unsigned char smem[32 * KPU * 2 + 32 * KP2 * 2];
  __shared__ float mu[32], rs[32];
  _Float16* ut  = (_Float16*)smem;                 // [32][512] f16
  float*    rb  = (float*)smem;                    // [32][256] f32 (overlay of ut)
  _Float16* y1  = (_Float16*)(smem + 32 * KPU * 2);// [32][256] f16
  const int t = threadIdx.x;
  const int node0 = blockIdx.x * 32;
  {  // stage u = [h | aggr] as f16
    const int nl = t >> 3, sub = t & 7;
    const int node = node0 + nl;
    const uint4* hp = (const uint4*)(hf16 + (size_t)node * H_);
    uint4* ur = (uint4*)(ut + nl * KPU);
    #pragma unroll
    for (int c = sub; c < 32; c += 8) cp16_async(&ur[c], &hp[c]);
    const float* ap = aggr + (size_t)node * H_;
    _Float16* u2 = ut + nl * KPU + H_;
    for (int c = sub * 32; c < sub * 32 + 32; ++c) u2[c] = (_Float16)ap[c];
  }
  async_join();
  __syncthreads();

  const int lane = t & 31, n0 = (t >> 5) * 32;
  const int lr = lane & 15, ksel = lane >> 4;
  __builtin_prefetch(Wu2 + (size_t)(n0 + lr) * KP2, 0, 1);

  v8f acc[2][2] = {};
  for (int kc = 0; kc < KPU / 32; ++kc) {
    const int kb = kc * 32 + ksel * 8;
    v16h bf0 = frag_ld(Wu1 + (size_t)(n0 + lr) * KPU + kb);
    v16h bf1 = frag_ld(Wu1 + (size_t)(n0 + 16 + lr) * KPU + kb);
    v16h a0 = frag_ld(ut + (size_t)lr * KPU + kb);
    v16h a1 = frag_ld(ut + (size_t)(16 + lr) * KPU + kb);
    acc[0][0] = wmma16(a0, bf0, acc[0][0]);
    acc[0][1] = wmma16(a0, bf1, acc[0][1]);
    acc[1][0] = wmma16(a1, bf0, acc[1][0]);
    acc[1][1] = wmma16(a1, bf1, acc[1][1]);
  }
  #pragma unroll
  for (int rt = 0; rt < 2; ++rt)
    #pragma unroll
    for (int f = 0; f < 2; ++f) {
      const int col = n0 + f * 16 + lr;
      const float bb = bu1[col];
      #pragma unroll
      for (int j = 0; j < 8; ++j) {
        const int row = rt * 16 + j + (ksel ? 8 : 0);
        y1[row * KP2 + col] = (_Float16)siluf(acc[rt][f][j] + bb);
      }
    }
  __syncthreads();

  v8f m[2][2] = {};
  for (int kc = 0; kc < KP2 / 32; ++kc) {
    const int kb = kc * 32 + ksel * 8;
    v16h bf0 = frag_ld(Wu2 + (size_t)(n0 + lr) * KP2 + kb);
    v16h bf1 = frag_ld(Wu2 + (size_t)(n0 + 16 + lr) * KP2 + kb);
    v16h a0 = frag_ld(y1 + (size_t)lr * KP2 + kb);
    v16h a1 = frag_ld(y1 + (size_t)(16 + lr) * KP2 + kb);
    m[0][0] = wmma16(a0, bf0, m[0][0]);
    m[0][1] = wmma16(a0, bf1, m[0][1]);
    m[1][0] = wmma16(a1, bf0, m[1][0]);
    m[1][1] = wmma16(a1, bf1, m[1][1]);
  }
  // residual into rb (ut no longer read after the y1 barrier -> safe overlay)
  #pragma unroll
  for (int rt = 0; rt < 2; ++rt)
    #pragma unroll
    for (int f = 0; f < 2; ++f) {
      const int col = n0 + f * 16 + lr;
      const float bb = bu2[col];
      #pragma unroll
      for (int j = 0; j < 8; ++j) {
        const int row = rt * 16 + j + (ksel ? 8 : 0);
        rb[row * H_ + col] =
            hbuf[(size_t)(node0 + row) * H_ + col] + m[rt][f][j] + bb;
      }
    }
  __syncthreads();
  if (t < 32) {
    float s = 0.f, s2 = 0.f;
    for (int c = 0; c < H_; ++c) { float v = rb[t * H_ + c]; s += v; s2 += v * v; }
    float mm = s * (1.0f / H_);
    mu[t] = mm;
    rs[t] = rsqrtf(fmaxf(s2 * (1.0f / H_) - mm * mm, 0.0f) + LN_EPS);
  }
  __syncthreads();
  #pragma unroll
  for (int rt = 0; rt < 2; ++rt)
    #pragma unroll
    for (int f = 0; f < 2; ++f) {
      const int col = n0 + f * 16 + lr;
      #pragma unroll
      for (int j = 0; j < 8; ++j) {
        const int row = rt * 16 + j + (ksel ? 8 : 0);
        const int node = node0 + row;
        float o = (rb[row * H_ + col] - mu[row]) * rs[row] * lnS[col] + lnB[col];
        o *= mask[node] ? 1.0f : 0.0f;
        hbuf[(size_t)node * H_ + col] = o;
        hf16[(size_t)node * H_ + col] = (_Float16)o;
        aggr[(size_t)node * H_ + col] = 0.0f;  // ready for next layer
      }
    }
}

// ------------------------- final LN + output linear ------------------------
__global__ void __launch_bounds__(256) out_k(const float* __restrict__ hbuf,
                                             const unsigned char* __restrict__ mask,
                                             const float* __restrict__ lnS,
                                             const float* __restrict__ lnB,
                                             const float* __restrict__ Wo,
                                             const float* __restrict__ bo,
                                             float* __restrict__ out) {
  __shared__ float row[H_];
  __shared__ float stat[2];
  const int flat = blockIdx.x, t = threadIdx.x;
  const float v = hbuf[(size_t)flat * H_ + t];
  row[t] = v;
  __syncthreads();
  if (t == 0) {
    float s = 0.f, s2 = 0.f;
    for (int c = 0; c < H_; ++c) { float x = row[c]; s += x; s2 += x * x; }
    float mm = s * (1.0f / H_);
    stat[0] = mm;
    stat[1] = rsqrtf(fmaxf(s2 * (1.0f / H_) - mm * mm, 0.0f) + LN_EPS);
  }
  __syncthreads();
  const float xln = (v - stat[0]) * stat[1] * lnS[t] + lnB[t];
  __syncthreads();
  row[t] = xln;
  __syncthreads();
  float o = bo[t];
  for (int c = 0; c < H_; ++c) o += row[c] * Wo[(size_t)c * H_ + t];
  o *= mask[flat] ? 1.0f : 0.0f;
  out[(size_t)flat * H_ + t] = o;
}

// ------------------------------- launcher ----------------------------------
extern "C" void kernel_launch(void* const* d_in, const int* in_sizes, int n_in,
                              void* d_out, int out_size, void* d_ws, size_t ws_size,
                              hipStream_t stream) {
  (void)in_sizes; (void)n_in; (void)out_size; (void)ws_size;
  const float* coords = (const float*)d_in[0];
  const unsigned char* mask = (const unsigned char*)d_in[1];
  const float* Wa = (const float*)d_in[2];
  const float* ba = (const float*)d_in[3];
  const float* Wp = (const float*)d_in[4];
  const float* bp = (const float*)d_in[5];
  const float* olnS = (const float*)d_in[36];
  const float* olnB = (const float*)d_in[37];
  const float* Wo   = (const float*)d_in[38];
  const float* bo   = (const float*)d_in[39];

  // workspace layout (bytes); all 16B aligned
  char* w = (char*)d_ws;
  _Float16* hf16   = (_Float16*)(w + 0);            // 4,194,304
  float*    aggr   = (float*)(w + 4194304);          // 8,388,608
  float*    hbuf   = (float*)(w + 12582912);         // 8,388,608
  int*      nbrf   = (int*)(w + 20971520);           //   983,040
  _Float16* ea16   = (_Float16*)(w + 21954560);      // 7,864,320
  float*    rbfm   = (float*)(w + 29818880);         //   524,288
  char*     wts    = w + 30343168;                   // f16 transposed weights
  const size_t LSTRIDE = 802816;                     // per-layer weight bytes
  const size_t OFF_WT1 = 0, OFF_WT2 = 278528, OFF_WU1 = 409600, OFF_WU2 = 671744;

  float* outf = (float*)d_out;
  int* esrc = (int*)(outf + (size_t)NNODES * H_);
  int* edst = esrc + NEDGES;

  // zero the scatter accumulator
  zerof<<<(NNODES * H_) / 256, 256, 0, stream>>>(aggr);

  // convert + transpose weights to f16 [N][Kpad]
  for (int i = 0; i < NL_; ++i) {
    const int li = 6 + i * 10;
    _Float16* wt1 = (_Float16*)(wts + (size_t)i * LSTRIDE + OFF_WT1);
    _Float16* wt2 = (_Float16*)(wts + (size_t)i * LSTRIDE + OFF_WT2);
    _Float16* wu1 = (_Float16*)(wts + (size_t)i * LSTRIDE + OFF_WU1);
    _Float16* wu2 = (_Float16*)(wts + (size_t)i * LSTRIDE + OFF_WU2);
    wtrans<<<KP1, 256, 0, stream>>>((const float*)d_in[li + 0], wt1, 2 * H_ + NRBF, KP1);
    wtrans<<<KP2, 256, 0, stream>>>((const float*)d_in[li + 2], wt2, H_, KP2);
    wtrans<<<KPU, 256, 0, stream>>>((const float*)d_in[li + 4], wu1, 2 * H_, KPU);
    wtrans<<<KP2, 256, 0, stream>>>((const float*)d_in[li + 6], wu2, H_, KP2);
  }

  // kNN graph + edge features + edge_index output
  knn<<<NNODES, 32, 0, stream>>>(coords, mask, nbrf, rbfm, ea16, esrc, edst);

  // initial node embedding
  init_h<<<NNODES, 256, 0, stream>>>(coords, mask, Wa, ba, Wp, bp, rbfm, hbuf, hf16);

  // GNN layers
  const size_t edge_lds = (size_t)MT * KP1 * 2 + (size_t)MT * KP2 * 2 + MT * 4;
  for (int i = 0; i < NL_; ++i) {
    const int li = 6 + i * 10;
    const _Float16* wt1 = (const _Float16*)(wts + (size_t)i * LSTRIDE + OFF_WT1);
    const _Float16* wt2 = (const _Float16*)(wts + (size_t)i * LSTRIDE + OFF_WT2);
    const _Float16* wu1 = (const _Float16*)(wts + (size_t)i * LSTRIDE + OFF_WU1);
    const _Float16* wu2 = (const _Float16*)(wts + (size_t)i * LSTRIDE + OFF_WU2);
    edge_msg<<<NEDGES / MT, 256, edge_lds, stream>>>(
        hf16, ea16, nbrf, wt1, (const float*)d_in[li + 1], wt2,
        (const float*)d_in[li + 3], aggr);
    node_update<<<NNODES / 32, 256, 0, stream>>>(
        hbuf, hf16, aggr, wu1, (const float*)d_in[li + 5], wu2,
        (const float*)d_in[li + 7], (const float*)d_in[li + 8],
        (const float*)d_in[li + 9], mask);
  }

  // output head
  out_k<<<NNODES, 256, 0, stream>>>(hbuf, mask, olnS, olnB, Wo, bo, outf);
}
